// EncoderRNN_90683939488320
// MI455X (gfx1250) — compile-verified, matrix operand order checked by reading, not run on hardware
//
#include <hip/hip_runtime.h>
#include <math.h>

// Problem sizes (fixed by the reference)
#define BSZ   256
#define TT    256
#define TM1   255
#define NN    256
#define HH    256
#define G4H   1024   // 4*H

typedef __attribute__((ext_vector_type(16))) _Float16 v16h;
typedef __attribute__((ext_vector_type(4)))  _Float16 v4h;
typedef __attribute__((ext_vector_type(8)))  float    v8f;
typedef __attribute__((ext_vector_type(4)))  float    v4f;

__device__ __forceinline__ float sigm(float x) { return 1.0f / (1.0f + __expf(-x)); }

// ---------------------------------------------------------------------------
// Kernel 1: attn[b,n] = softmax_n( sum_t x[b,t,n] * w_x[t] )
// (the per-batch scalar (h·w_h + c·w_c) and b_a cancel inside softmax)
// ---------------------------------------------------------------------------
__global__ void __launch_bounds__(256) attn_kernel(const float* __restrict__ x,
                                                   const float* __restrict__ W_a,
                                                   float* __restrict__ attn) {
    __shared__ float red[256];
    const int b = blockIdx.x;
    const int n = threadIdx.x;
    const float* wx = W_a + 2 * HH;               // w_x, length T-1
    const float* xp = x + (size_t)b * TM1 * NN + n;
    float s = 0.0f;
    for (int t = 0; t < TM1; ++t) s += xp[(size_t)t * NN] * wx[t];
    red[n] = s; __syncthreads();
    for (int off = 128; off > 0; off >>= 1) {
        if (n < off) red[n] = fmaxf(red[n], red[n + off]);
        __syncthreads();
    }
    const float mx = red[0]; __syncthreads();
    const float e = __expf(s - mx);
    red[n] = e; __syncthreads();
    for (int off = 128; off > 0; off >>= 1) {
        if (n < off) red[n] += red[n + off];
        __syncthreads();
    }
    attn[(size_t)b * NN + n] = e / red[0];
}

// ---------------------------------------------------------------------------
// Kernel 2: pack W_ih/W_hh to f16, bias = b_ih + b_hh, zero h buffer 0,
//           reset the grid barrier. 262144 threads.
// ---------------------------------------------------------------------------
__global__ void __launch_bounds__(256) pack_kernel(const float* __restrict__ W_ih,
                                                   const float* __restrict__ W_hh,
                                                   const float* __restrict__ b_ih,
                                                   const float* __restrict__ b_hh,
                                                   _Float16* __restrict__ wihh,
                                                   _Float16* __restrict__ whhh,
                                                   float* __restrict__ bias,
                                                   _Float16* __restrict__ hbf,
                                                   unsigned* __restrict__ bar) {
    const int idx = blockIdx.x * 256 + threadIdx.x;   // 0 .. 262143
    wihh[idx] = (_Float16)W_ih[idx];
    whhh[idx] = (_Float16)W_hh[idx];
    if (idx < G4H)       bias[idx] = b_ih[idx] + b_hh[idx];
    if (idx < BSZ * HH)  hbf[idx]  = (_Float16)0.0f;  // h_0 = 0 (buffer 0)
    if (idx == 0) { bar[0] = 0u; bar[1] = 0u; }
}

// ---------------------------------------------------------------------------
// Kernel 3: weighted[b,t,n] = attn[b,n]*x[b,t,n]  -> d_out half 1 (f32, b-major)
//           plus an f16 copy in (t, b, n) order for the WMMA GEMM A-operand
//           (GEMM rows r = t*256+b decode with shifts). float4-vectorized.
// ---------------------------------------------------------------------------
__global__ void __launch_bounds__(256) weighted_kernel(const float* __restrict__ x,
                                                       const float* __restrict__ attn,
                                                       float* __restrict__ outw,
                                                       _Float16* __restrict__ wbf) {
    const unsigned idx4 = blockIdx.x * 256 + threadIdx.x;   // 0 .. 4,177,919
    const size_t base = (size_t)idx4 * 4;
    const unsigned b = (unsigned)(base / (TM1 * NN));
    const unsigned rem = (unsigned)(base - (size_t)b * (TM1 * NN));
    const unsigned t = rem / NN;
    const unsigned n = rem & (NN - 1);

    const v4f xv = *(const v4f*)(x + base);
    const v4f av = *(const v4f*)(attn + (size_t)b * NN + n);
    v4f wv;
    v4h wh;
#pragma unroll
    for (int i = 0; i < 4; ++i) {
        wv[i] = av[i] * xv[i];
        wh[i] = (_Float16)wv[i];
    }
    *(v4f*)(outw + base) = wv;
    *(v4h*)(wbf + ((size_t)t * BSZ + b) * NN + n) = wh;
}

// ---------------------------------------------------------------------------
// Kernel 4: xgates = weighted_f16 @ W_ih^T + (b_ih+b_hh), f32 accumulate,
//           stored as f16 (halves the dominant 267MB intermediate).
// M = (T-1)*B = 65280 rows (r = t*256+b), N = 1024, K = 256.
// One 16x16 tile per wave, v_wmma_f32_16x16x32_f16, 8 K-steps.
// Output layout: xg[t][gate][b][h]  (coalesced reads in the recurrence).
// ---------------------------------------------------------------------------
__global__ void __launch_bounds__(256) xgates_kernel(const _Float16* __restrict__ wbf,
                                                     const _Float16* __restrict__ wihh,
                                                     const float* __restrict__ bias,
                                                     _Float16* __restrict__ xg) {
    const int wave = blockIdx.x * 8 + (threadIdx.x >> 5);
    const int lane = threadIdx.x & 31;
    const int rowT = wave >> 6;          // 0..4079 (16 rows, same t)
    const int colT = wave & 63;          // 0..63
    const int nrel = lane & 15;
    const int khi  = (lane >> 4) * 16;   // upper half-wave holds K+16..K+31
    const int mhi  = (lane >> 4) * 8;    // C/D: lanes 16-31 hold M=i+8

    const _Float16* ar = wbf  + (size_t)(rowT * 16 + nrel) * NN + khi;
    const _Float16* br = wihh + (size_t)(colT * 16 + nrel) * NN + khi;

    v8f acc = {};
#pragma unroll
    for (int k0 = 0; k0 < NN; k0 += 32) {
        v16h a = *(const v16h*)(ar + k0);
        v16h b = *(const v16h*)(br + k0);
        acc = __builtin_amdgcn_wmma_f32_16x16x32_f16(false, a, false, b,
                                                     (short)0, acc, false, false);
    }
    const int j    = colT * 16 + nrel;   // 0..1023
    const int gate = j >> 8;
    const int hc   = j & 255;
    const int r0   = rowT * 16;
    const int t    = r0 >> 8;            // all 16 rows share t
    const int b0   = r0 & 255;
    const float bv = bias[j];
#pragma unroll
    for (int i = 0; i < 8; ++i) {
        const int bb = b0 + i + mhi;
        xg[(((size_t)t * 4 + gate) * BSZ + bb) * HH + hc] = (_Float16)(acc[i] + bv);
    }
}

// ---------------------------------------------------------------------------
// Device-wide barrier for the persistent recurrence kernel.
// ---------------------------------------------------------------------------
__device__ __forceinline__ void grid_sync(unsigned* cnt, unsigned* gen, unsigned nb) {
    __syncthreads();
    if (threadIdx.x == 0) {
        const unsigned g = __hip_atomic_load(gen, __ATOMIC_RELAXED, __HIP_MEMORY_SCOPE_AGENT);
        const unsigned v = __hip_atomic_fetch_add(cnt, 1u, __ATOMIC_ACQ_REL, __HIP_MEMORY_SCOPE_AGENT);
        if (v == nb - 1u) {
            __hip_atomic_store(cnt, 0u, __ATOMIC_RELAXED, __HIP_MEMORY_SCOPE_AGENT);
            __hip_atomic_fetch_add(gen, 1u, __ATOMIC_ACQ_REL, __HIP_MEMORY_SCOPE_AGENT);
        } else {
            while (__hip_atomic_load(gen, __ATOMIC_ACQUIRE, __HIP_MEMORY_SCOPE_AGENT) == g)
                __builtin_amdgcn_s_sleep(1);
        }
    }
    __syncthreads();
}

// ---------------------------------------------------------------------------
// Kernel 5: persistent LSTM recurrence. 128 blocks x 64 threads = 256 waves.
// Wave (rowT, colT) owns batch rows m0..m0+15 and hidden cols n0..n0+15.
//  - ALL 32 W_hh B-fragments (4 gates x 8 k-steps) live in VGPRs across the
//    whole t-loop (~300 VGPRs/wave; occupancy irrelevant at 256 waves), so the
//    255-step sequential chain only loads the h-dependent A tiles.
//  - xg (f16) for step t+1 is prefetched into registers BEFORE the barrier,
//    hiding its latency behind the barrier + WMMA phase.
//  - c state lives in registers; h is double-buffered f16 in global.
// ---------------------------------------------------------------------------
__global__ void __launch_bounds__(64) rec_kernel(const _Float16* __restrict__ xg,
                                                 const _Float16* __restrict__ whhh,
                                                 _Float16* __restrict__ hbf,
                                                 float* __restrict__ out_enc,
                                                 unsigned* __restrict__ bar) {
    const int wave = (blockIdx.x << 1) + (threadIdx.x >> 5);  // 0..255
    const int lane = threadIdx.x & 31;
    const int m0   = (wave >> 4) << 4;   // batch-row tile base
    const int n0   = (wave & 15) << 4;   // hidden-col tile base
    const int nrel = lane & 15;
    const int khi  = (lane >> 4) * 16;
    const int mhi  = (lane >> 4) * 8;
    const int hc   = n0 + nrel;
    unsigned* cnt = bar;
    unsigned* gen = bar + 1;

    // ---- hoist all W_hh B fragments into registers (loop-invariant) ----
    v16h Bt[4][8];
#pragma unroll
    for (int g = 0; g < 4; ++g) {
        const _Float16* bp = whhh + (size_t)(g * 256 + n0 + nrel) * HH + khi;
#pragma unroll
        for (int kk = 0; kk < 8; ++kk)
            Bt[g][kk] = *(const v16h*)(bp + kk * 32);
    }

    float cst[8];
#pragma unroll
    for (int i = 0; i < 8; ++i) cst[i] = 0.0f;

    // ---- prefetch xg for t=0 ----
    float xgr[4][8];
#pragma unroll
    for (int g = 0; g < 4; ++g)
#pragma unroll
        for (int i = 0; i < 8; ++i)
            xgr[g][i] = (float)xg[((size_t)g * BSZ + (m0 + i + mhi)) * HH + hc];

    for (int t = 0; t < TM1; ++t) {
        const _Float16* hr = hbf + (size_t)(t & 1) * (BSZ * HH);        // h_t
        _Float16*       hw = hbf + (size_t)((t + 1) & 1) * (BSZ * HH);  // h_{t+1}
        const _Float16* ar = hr + (size_t)(m0 + nrel) * HH + khi;

        v8f ai = {}, af = {}, ag = {}, ao = {};
#pragma unroll
        for (int kk = 0; kk < 8; ++kk) {
            v16h a = *(const v16h*)(ar + kk * 32);
            ai = __builtin_amdgcn_wmma_f32_16x16x32_f16(false, a, false, Bt[0][kk], (short)0, ai, false, false);
            af = __builtin_amdgcn_wmma_f32_16x16x32_f16(false, a, false, Bt[1][kk], (short)0, af, false, false);
            ag = __builtin_amdgcn_wmma_f32_16x16x32_f16(false, a, false, Bt[2][kk], (short)0, ag, false, false);
            ao = __builtin_amdgcn_wmma_f32_16x16x32_f16(false, a, false, Bt[3][kk], (short)0, ao, false, false);
        }

        // ---- LSTM cell (c in registers), write h_{t+1} ----
#pragma unroll
        for (int i = 0; i < 8; ++i) {
            const int bb = m0 + i + mhi;                       // batch row
            const float gi = ai[i] + xgr[0][i];
            const float gf = af[i] + xgr[1][i];
            const float gg = ag[i] + xgr[2][i];
            const float go = ao[i] + xgr[3][i];
            const float cn = sigm(gf) * cst[i] + sigm(gi) * tanhf(gg);
            const float hn = sigm(go) * tanhf(cn);
            cst[i] = cn;
            out_enc[((size_t)bb * TM1 + t) * HH + hc] = hn;
            hw[(size_t)bb * HH + hc] = (_Float16)hn;
        }

        // ---- prefetch xg for t+1 (overlaps the barrier latency) ----
        const int tn = (t + 1 < TM1) ? (t + 1) : t;
#pragma unroll
        for (int g = 0; g < 4; ++g)
#pragma unroll
            for (int i = 0; i < 8; ++i)
                xgr[g][i] = (float)xg[(((size_t)tn * 4 + g) * BSZ + (m0 + i + mhi)) * HH + hc];

        grid_sync(cnt, gen, gridDim.x);
    }
}

// ---------------------------------------------------------------------------
// Host launcher
// ---------------------------------------------------------------------------
extern "C" void kernel_launch(void* const* d_in, const int* in_sizes, int n_in,
                              void* d_out, int out_size, void* d_ws, size_t ws_size,
                              hipStream_t stream) {
    (void)in_sizes; (void)n_in; (void)out_size; (void)ws_size;
    const float* x    = (const float*)d_in[0];  // (B, T-1, N)
    // d_in[1] = hidden (unused: h0/c0 are zeros)
    const float* W_a  = (const float*)d_in[2];
    // d_in[3] = b_a (cancels in softmax)
    const float* W_ih = (const float*)d_in[4];
    const float* W_hh = (const float*)d_in[5];
    const float* b_ih = (const float*)d_in[6];
    const float* b_hh = (const float*)d_in[7];

    float* out_w = (float*)d_out;                                 // (B, T-1, N)
    float* out_e = out_w + (size_t)BSZ * TM1 * NN;                // (B, T-1, H)

    // Workspace layout (256B-aligned sections), total ~169MB
    char* ws = (char*)d_ws;
    float*     attn = (float*)    (ws + 0);               // 256KB
    float*     bias = (float*)    (ws + 262144);          // 4KB
    _Float16*  wihh = (_Float16*) (ws + 266240);          // 512KB
    _Float16*  whhh = (_Float16*) (ws + 790528);          // 512KB
    _Float16*  hbf  = (_Float16*) (ws + 1314816);         // 256KB (double buffer)
    unsigned*  bar  = (unsigned*) (ws + 1576960);         // barrier
    _Float16*  wbf  = (_Float16*) (ws + 1577216);         // 33.4MB weighted f16 (t,b,n)
    _Float16*  xg   = (_Float16*) (ws + 35000576);        // 133.7MB xgates f16 [t][g][b][h]

    attn_kernel    <<<BSZ,   256, 0, stream>>>(x, W_a, attn);
    pack_kernel    <<<1024,  256, 0, stream>>>(W_ih, W_hh, b_ih, b_hh, wihh, whhh, bias, hbf, bar);
    weighted_kernel<<<16320, 256, 0, stream>>>(x, attn, out_w, wbf);
    xgates_kernel  <<<32640, 256, 0, stream>>>(wbf, wihh, bias, xg);
    rec_kernel     <<<128,   64,  0, stream>>>(xg, whhh, hbf, out_e, bar);
}
